// SparseMLP_21861383537336
// MI455X (gfx1250) — compile-verified, matrix operand order checked by reading, not run on hardware
//
#include <hip/hip_runtime.h>
#include <hip/hip_bf16.h>

#define NUM_EXPERTS 16
#define TOPK 4
#define HIDDEN 1024
#define INTER 1024
#define GU (2 * INTER)
#define T_TOKENS 4096
#define ALPHA 1.702f
#define LIMIT 7.0f

typedef __attribute__((ext_vector_type(16))) _Float16 v16h;
typedef __attribute__((ext_vector_type(2)))  _Float16 h2;
typedef __attribute__((ext_vector_type(8)))  float    v8f;
typedef __attribute__((ext_vector_type(4)))  int      v4i;

#define KT 32     // K tile = one WMMA step
#define LDK 34    // padded LDS stride in f16 (17-bank step -> conflict free)
#define LDKA 40   // down-A stride: 80B rows, 16B aligned for async B128 copies

#if defined(__AMDGCN__) &&                                         \
    __has_builtin(__builtin_amdgcn_global_load_async_to_lds_b128) && \
    __has_builtin(__builtin_amdgcn_s_wait_asynccnt)
#define HAVE_ASYNC_LDS 1
typedef v4i __attribute__((address_space(1)))* gptr_v4i;
typedef v4i __attribute__((address_space(3)))* lptr_v4i;
#else
#define HAVE_ASYNC_LDS 0
#endif

// ---------------------------------------------------------------- router ----
__global__ __launch_bounds__(128) void moe_router_kernel(
    const float* __restrict__ x, const float* __restrict__ rw,
    const float* __restrict__ rb, float* __restrict__ out_scores,
    int* __restrict__ topk_idx, float* __restrict__ topk_w) {
  __shared__ float xs[8 * HIDDEN];
  __shared__ float lg[8 * NUM_EXPERTS];
  const int tid = threadIdx.x;
  const int tok0 = blockIdx.x * 8;
  for (int j = 0; j < 64; ++j) {
    int idx = tid + j * 128;
    xs[idx] = x[(size_t)tok0 * HIDDEN + idx];
  }
  __syncthreads();
  const int t = tid >> 4, e = tid & 15;
  const float* w = rw + e * HIDDEN;
  const float* xv = xs + t * HIDDEN;
  float acc = 0.f;
#pragma unroll 8
  for (int h = 0; h < HIDDEN; ++h) acc = fmaf(xv[h], w[h], acc);
  lg[t * NUM_EXPERTS + e] = acc + rb[e];
  __syncthreads();
  if (tid < 8) {
    float v[NUM_EXPERTS];
    for (int i = 0; i < NUM_EXPERTS; ++i) v[i] = lg[tid * NUM_EXPERTS + i];
    int idxs[TOPK];
    float vals[TOPK];
    for (int k = 0; k < TOPK; ++k) {
      int bi = 0;
      float bv = v[0];
      for (int i = 1; i < NUM_EXPERTS; ++i)
        if (v[i] > bv) { bv = v[i]; bi = i; }
      idxs[k] = bi; vals[k] = bv; v[bi] = -1e30f;
    }
    const float m = vals[0];
    float s = 0.f, ex[TOPK];
    for (int k = 0; k < TOPK; ++k) { ex[k] = __expf(vals[k] - m); s += ex[k]; }
    float scores[NUM_EXPERTS];
    for (int i = 0; i < NUM_EXPERTS; ++i) scores[i] = 0.f;
    const int tok = tok0 + tid;
    for (int k = 0; k < TOPK; ++k) {
      float wk = ex[k] / s;
      scores[idxs[k]] = wk;
      topk_idx[tok * TOPK + k] = idxs[k];
      topk_w[tok * TOPK + k] = wk;
    }
    for (int i = 0; i < NUM_EXPERTS; ++i)
      out_scores[(size_t)tok * NUM_EXPERTS + i] = scores[i];
  }
}

// ----------------------------------------------- deterministic scatter ------
__global__ __launch_bounds__(256) void moe_scatter_kernel(
    const int* __restrict__ topk_idx, const float* __restrict__ topk_w,
    int* __restrict__ list, float* __restrict__ wlist, int* __restrict__ counts) {
  __shared__ int cnt_s[256];
  const int e = blockIdx.x, tid = threadIdx.x;
  const int t0 = tid * (T_TOKENS / 256);  // 16 tokens/thread
  int c = 0;
  for (int t = t0; t < t0 + T_TOKENS / 256; ++t)
    for (int k = 0; k < TOPK; ++k)
      if (topk_idx[t * TOPK + k] == e) ++c;
  cnt_s[tid] = c;
  __syncthreads();
  for (int off = 1; off < 256; off <<= 1) {
    int v = (tid >= off) ? cnt_s[tid - off] : 0;
    __syncthreads();
    cnt_s[tid] += v;
    __syncthreads();
  }
  int base = cnt_s[tid] - c;  // exclusive prefix
  for (int t = t0; t < t0 + T_TOKENS / 256; ++t)
    for (int k = 0; k < TOPK; ++k)
      if (topk_idx[t * TOPK + k] == e) {
        list[e * T_TOKENS + base] = t;
        wlist[e * T_TOKENS + base] = topk_w[t * TOPK + k];
        ++base;
      }
  if (tid == 255) counts[e] = cnt_s[255];
}

__global__ void moe_offsets_kernel(const int* __restrict__ counts,
                                   int* __restrict__ offsets) {
  if (threadIdx.x == 0) {
    int s = 0;
    for (int e = 0; e < NUM_EXPERTS; ++e) { offsets[e] = s; s += counts[e]; }
    offsets[NUM_EXPERTS] = s;
  }
}

// -------------------------------------------------- out = sum_k w_k*bias ----
__global__ __launch_bounds__(256) void moe_outbias_kernel(
    const int* __restrict__ topk_idx, const float* __restrict__ topk_w,
    const float* __restrict__ dbias, float* __restrict__ out) {
  const int idx = blockIdx.x * 256 + threadIdx.x;
  const int t = idx >> 10, h = idx & 1023;
  float s = 0.f;
#pragma unroll
  for (int k = 0; k < TOPK; ++k)
    s = fmaf(topk_w[t * TOPK + k], dbias[topk_idx[t * TOPK + k] * HIDDEN + h], s);
  out[idx] = s;
}

// --------------------------------------------------------- WMMA helpers -----
union Frag16 { v16h v; unsigned int u[8]; };

// rc = (16-aligned base) + (lane&15); K offset from lane>>4 per ISA layout
template <int STRIDE>
__device__ __forceinline__ v16h load_frag(const _Float16* lds, int rc, int lane) {
  const _Float16* p = lds + rc * STRIDE + ((lane >> 4) << 3);
  Frag16 f;
#pragma unroll
  for (int j = 0; j < 4; ++j) {
    f.u[j]     = *(const unsigned int*)(p + 2 * j);
    f.u[4 + j] = *(const unsigned int*)(p + 16 + 2 * j);
  }
  return f.v;
}

// ---------------------------------------------- gate_up grouped GEMM --------
__global__ __launch_bounds__(256) void moe_gateup_kernel(
    const float* __restrict__ x, const float* __restrict__ gup,
    const float* __restrict__ gup_bias, const int* __restrict__ list,
    const int* __restrict__ counts, const int* __restrict__ offsets,
    _Float16* __restrict__ act) {
  const int e = blockIdx.z;
  const int cnt = counts[e];
  const int m0 = blockIdx.y * 128;
  if (m0 >= cnt) return;
  const int n0 = blockIdx.x * 128;  // over 2048 gate_up cols

  __shared__ _Float16 ldsA[2][128 * LDK];
  __shared__ _Float16 ldsB[2][128 * LDK];
  const int tid = threadIdx.x;
  const int lane = tid & 31, wid = tid >> 5;
  const int wm = wid >> 1, wn = wid & 1;  // 4x2 wave grid, each 32x64

  const int* elist = list + e * T_TOKENS;

  // hoisted gather setup: rows fixed per thread across the K loop
  const int kp = tid & 15;        // f16-pair index within K tile (A)
  const int ra0 = tid >> 4;       // rows ra0 + 16j, j<8
  const float* aSrc[8];
  bool aValid[8];
#pragma unroll
  for (int j = 0; j < 8; ++j) {
    int row = m0 + ra0 + 16 * j;
    aValid[j] = (row < cnt);
    int tok = aValid[j] ? elist[row] : 0;
    aSrc[j] = x + (size_t)tok * HIDDEN + 2 * kp;
  }
  const int bn = (tid & 31) * 4;  // 4 consecutive cols per thread
  const int bk0 = tid >> 5;       // K rows bk0 + 8j, j<4
  const float* W = gup + (size_t)e * HIDDEN * GU + n0 + bn;

  float2 ga[8];
  float4 gb[4];

  v8f acc[2][4];
#pragma unroll
  for (int m = 0; m < 2; ++m)
#pragma unroll
    for (int n = 0; n < 4; ++n)
#pragma unroll
      for (int r = 0; r < 8; ++r) acc[m][n][r] = 0.f;

  auto load_tiles = [&](int k0) {
#pragma unroll
    for (int j = 0; j < 8; ++j)
      ga[j] = aValid[j] ? *(const float2*)(aSrc[j] + k0) : make_float2(0.f, 0.f);
#pragma unroll
    for (int j = 0; j < 4; ++j)
      gb[j] = *(const float4*)(W + (size_t)(k0 + bk0 + 8 * j) * GU);
  };
  auto store_tiles = [&](int buf) {
#pragma unroll
    for (int j = 0; j < 8; ++j) {
      h2 p = {(_Float16)ga[j].x, (_Float16)ga[j].y};
      *(h2*)&ldsA[buf][(ra0 + 16 * j) * LDK + 2 * kp] = p;
    }
#pragma unroll
    for (int j = 0; j < 4; ++j) {
      int kk = bk0 + 8 * j;
      ldsB[buf][(bn + 0) * LDK + kk] = (_Float16)gb[j].x;
      ldsB[buf][(bn + 1) * LDK + kk] = (_Float16)gb[j].y;
      ldsB[buf][(bn + 2) * LDK + kk] = (_Float16)gb[j].z;
      ldsB[buf][(bn + 3) * LDK + kk] = (_Float16)gb[j].w;
    }
  };

  load_tiles(0);
  store_tiles(0);
  int buf = 0;
  for (int k0 = 0; k0 < HIDDEN; k0 += KT) {
    __syncthreads();  // tile k0 visible; prior reads of buf^1 finished
    const bool more = (k0 + KT < HIDDEN);
    if (more) load_tiles(k0 + KT);  // globals in flight during WMMA

    v16h a[2], b[4];
#pragma unroll
    for (int m = 0; m < 2; ++m)
      a[m] = load_frag<LDK>(ldsA[buf], wm * 32 + m * 16 + (lane & 15), lane);
#pragma unroll
    for (int n = 0; n < 4; ++n)
      b[n] = load_frag<LDK>(ldsB[buf], wn * 64 + n * 16 + (lane & 15), lane);
#pragma unroll
    for (int m = 0; m < 2; ++m)
#pragma unroll
      for (int n = 0; n < 4; ++n)
        acc[m][n] = __builtin_amdgcn_wmma_f32_16x16x32_f16(
            false, a[m], false, b[n], (short)0, acc[m][n], false, false);

    if (more) {
      store_tiles(buf ^ 1);
      buf ^= 1;
    }
  }

  // epilogue: pair even(gate)/odd(up) columns via lane shuffle, GLU, store f16
  const int half = lane >> 4;
  const int lcol = lane & 15;
  const size_t ebase = (size_t)offsets[e];
  const bool evenlane = ((lcol & 1) == 0);
#pragma unroll
  for (int m = 0; m < 2; ++m)
#pragma unroll
    for (int n = 0; n < 4; ++n) {
      const int colg = n0 + wn * 64 + n * 16 + lcol;
      const float bias_g = gup_bias[e * GU + colg];
      const float bias_u = gup_bias[e * GU + (colg | 1)];
#pragma unroll
      for (int r = 0; r < 8; ++r) {
        float v = acc[m][n][r];
        float partner = __shfl_xor(v, 1, 32);
        if (evenlane) {
          int row = m0 + wm * 32 + m * 16 + r + 8 * half;
          if (row < cnt) {
            float g = fminf(v + bias_g, LIMIT);
            float u = fminf(fmaxf(partner + bias_u, -LIMIT), LIMIT);
            float glu = g / (1.f + __expf(-ALPHA * g));
            act[(ebase + row) * (size_t)INTER + (colg >> 1)] =
                (_Float16)((u + 1.f) * glu);
          }
        }
      }
    }
}

// ---------------------------------------------- down grouped GEMM -----------
__global__ __launch_bounds__(256) void moe_down_kernel(
    const _Float16* __restrict__ act, const float* __restrict__ dw,
    const int* __restrict__ list, const float* __restrict__ wlist,
    const int* __restrict__ counts, const int* __restrict__ offsets,
    float* __restrict__ out) {
  const int e = blockIdx.z;
  const int cnt = counts[e];
  const int m0 = blockIdx.y * 128;
  if (m0 >= cnt) return;
  const int n0 = blockIdx.x * 128;  // over HIDDEN

  __shared__ _Float16 ldsA[2][128 * LDKA];
  __shared__ _Float16 ldsB[2][128 * LDK];
  const int tid = threadIdx.x;
  const int lane = tid & 31, wid = tid >> 5;
  const int wm = wid >> 1, wn = wid & 1;

  const size_t ebase = (size_t)offsets[e];

  const int bn = (tid & 31) * 4;
  const int bk0 = tid >> 5;
  const float* W = dw + (size_t)e * INTER * HIDDEN + n0 + bn;

  float4 gb[4];
  auto loadB = [&](int k0) {
#pragma unroll
    for (int j = 0; j < 4; ++j)
      gb[j] = *(const float4*)(W + (size_t)(k0 + bk0 + 8 * j) * HIDDEN);
  };
  auto storeB = [&](int buf) {
#pragma unroll
    for (int j = 0; j < 4; ++j) {
      int kk = bk0 + 8 * j;
      ldsB[buf][(bn + 0) * LDK + kk] = (_Float16)gb[j].x;
      ldsB[buf][(bn + 1) * LDK + kk] = (_Float16)gb[j].y;
      ldsB[buf][(bn + 2) * LDK + kk] = (_Float16)gb[j].z;
      ldsB[buf][(bn + 3) * LDK + kk] = (_Float16)gb[j].w;
    }
  };

#if HAVE_ASYNC_LDS
  // async A path: each lane copies 16B (8 f16) straight global->LDS.
  // rows: (tid>>2) + 64j, j<2 ; kq = (tid&3)*8 halves (16B aligned w/ LDKA=40)
  const int arow = tid >> 2;
  const int akq = (tid & 3) * 8;
  // pre-zero rows that will never be async-written (tail guard)
#pragma unroll
  for (int j = 0; j < 2; ++j) {
    int row = arow + 64 * j;
    if (m0 + row >= cnt) {
      uint4 z = {0u, 0u, 0u, 0u};
      *(uint4*)&ldsA[0][row * LDKA + akq] = z;
      *(uint4*)&ldsA[1][row * LDKA + akq] = z;
    }
  }
  auto issueAsyncA = [&](int bufi, int k0) {
#pragma unroll
    for (int j = 0; j < 2; ++j) {
      int row = arow + 64 * j;
      if (m0 + row < cnt) {
        const _Float16* gp =
            act + (ebase + m0 + row) * (size_t)INTER + k0 + akq;
        __builtin_amdgcn_global_load_async_to_lds_b128(
            (gptr_v4i)(void*)gp,
            (lptr_v4i)(void*)&ldsA[bufi][row * LDKA + akq],
            0, 0);
      }
    }
  };
#else
  const int kp = tid & 15;   // f16-pair index (A, loaded as b32)
  const int ra0 = tid >> 4;  // rows ra0 + 16j, j<8
  const _Float16* aSrc[8];
  bool aValid[8];
#pragma unroll
  for (int j = 0; j < 8; ++j) {
    int row = m0 + ra0 + 16 * j;
    aValid[j] = (row < cnt);
    size_t r = aValid[j] ? (ebase + row) : ebase;
    aSrc[j] = act + r * (size_t)INTER + 2 * kp;
  }
  unsigned int ga[8];
  auto loadA = [&](int k0) {
#pragma unroll
    for (int j = 0; j < 8; ++j)
      ga[j] = aValid[j] ? *(const unsigned int*)(aSrc[j] + k0) : 0u;
  };
  auto storeA = [&](int buf) {
#pragma unroll
    for (int j = 0; j < 8; ++j)
      *(unsigned int*)&ldsA[buf][(ra0 + 16 * j) * LDKA + 2 * kp] = ga[j];
  };
#endif

  v8f acc[2][4];
#pragma unroll
  for (int m = 0; m < 2; ++m)
#pragma unroll
    for (int n = 0; n < 4; ++n)
#pragma unroll
      for (int r = 0; r < 8; ++r) acc[m][n][r] = 0.f;

  // prologue
#if HAVE_ASYNC_LDS
  issueAsyncA(0, 0);
#else
  loadA(0);
  storeA(0);
#endif
  loadB(0);
  storeB(0);

  int buf = 0;
  for (int k0 = 0; k0 < INTER; k0 += KT) {
#if HAVE_ASYNC_LDS
    __builtin_amdgcn_s_wait_asynccnt(0);  // this wave's async A writes landed
#endif
    __syncthreads();
    const bool more = (k0 + KT < INTER);
    if (more) {
#if HAVE_ASYNC_LDS
      issueAsyncA(buf ^ 1, k0 + KT);  // buf^1 readers passed the barrier above
#else
      loadA(k0 + KT);
#endif
      loadB(k0 + KT);
    }

    v16h a[2], b[4];
#pragma unroll
    for (int m = 0; m < 2; ++m)
      a[m] = load_frag<LDKA>(ldsA[buf], wm * 32 + m * 16 + (lane & 15), lane);
#pragma unroll
    for (int n = 0; n < 4; ++n)
      b[n] = load_frag<LDK>(ldsB[buf], wn * 64 + n * 16 + (lane & 15), lane);
#pragma unroll
    for (int m = 0; m < 2; ++m)
#pragma unroll
      for (int n = 0; n < 4; ++n)
        acc[m][n] = __builtin_amdgcn_wmma_f32_16x16x32_f16(
            false, a[m], false, b[n], (short)0, acc[m][n], false, false);

    if (more) {
#if !HAVE_ASYNC_LDS
      storeA(buf ^ 1);
#endif
      storeB(buf ^ 1);
      buf ^= 1;
    }
  }

  const int half = lane >> 4;
  const int lcol = lane & 15;
#pragma unroll
  for (int m = 0; m < 2; ++m)
#pragma unroll
    for (int n = 0; n < 4; ++n) {
      const int colg = n0 + wn * 64 + n * 16 + lcol;
#pragma unroll
      for (int r = 0; r < 8; ++r) {
        int row = m0 + wm * 32 + m * 16 + r + 8 * half;
        if (row < cnt) {
          int tok = list[e * T_TOKENS + row];
          float wk = wlist[e * T_TOKENS + row];
          atomicAdd(&out[(size_t)tok * HIDDEN + colg], acc[m][n][r] * wk);
        }
      }
    }
}

// ---------------------------------------------------------------- launch ----
extern "C" void kernel_launch(void* const* d_in, const int* in_sizes, int n_in,
                              void* d_out, int out_size, void* d_ws,
                              size_t ws_size, hipStream_t stream) {
  const float* x     = (const float*)d_in[0];
  const float* rw    = (const float*)d_in[1];
  const float* rb    = (const float*)d_in[2];
  const float* gup   = (const float*)d_in[3];
  const float* gup_b = (const float*)d_in[4];
  const float* dw    = (const float*)d_in[5];
  const float* db    = (const float*)d_in[6];

  float* out = (float*)d_out;
  float* out_scores = out + (size_t)T_TOKENS * HIDDEN;

  char* ws = (char*)d_ws;
  int*      topk_idx = (int*)(ws + 0);            //  64 KB
  float*    topk_w   = (float*)(ws + 65536);      //  64 KB
  int*      counts   = (int*)(ws + 131072);
  int*      offsets  = (int*)(ws + 131136);
  int*      list     = (int*)(ws + 262144);       // 256 KB
  float*    wlist    = (float*)(ws + 524288);     // 256 KB
  _Float16* act      = (_Float16*)(ws + 1048576); //  32 MB (4*T x INTER f16)

  moe_router_kernel<<<T_TOKENS / 8, 128, 0, stream>>>(x, rw, rb, out_scores,
                                                      topk_idx, topk_w);
  moe_scatter_kernel<<<NUM_EXPERTS, 256, 0, stream>>>(topk_idx, topk_w, list,
                                                      wlist, counts);
  moe_offsets_kernel<<<1, 32, 0, stream>>>(counts, offsets);
  moe_outbias_kernel<<<(T_TOKENS * HIDDEN) / 256, 256, 0, stream>>>(
      topk_idx, topk_w, db, out);

  dim3 gc(GU / 128, T_TOKENS / 128, NUM_EXPERTS);      // 16 x 32 x 16
  moe_gateup_kernel<<<gc, 256, 0, stream>>>(x, gup, gup_b, list, counts,
                                            offsets, act);
  dim3 gd(HIDDEN / 128, T_TOKENS / 128, NUM_EXPERTS);  // 8 x 32 x 16
  moe_down_kernel<<<gd, 256, 0, stream>>>(act, dw, list, wlist, counts,
                                          offsets, out);
}
